// GATModel_20040317403819
// MI455X (gfx1250) — matrix operand order repaired and checked by reading, not compiled, not run on hardware
//
#include <hip/hip_runtime.h>
#include <hip/hip_bf16.h>

// ---------------- problem constants (match reference) ----------------
#define NN      50000      // nodes
#define NE      800000     // edges
#define FIN     1024       // input feature dim
#define HID     512        // hidden dim
#define NG      8          // graphs
#define NCLS    6          // classes
#define NEGSLP  0.2f
#define EPSI    1e-16f

// encoded -inf for monotone float<->uint max trick: fenc(-inf) = ~0xFF800000
#define ENC_NEG_INF 0x007FFFFFu

typedef __attribute__((ext_vector_type(8)))  __bf16 v8bf;
typedef __attribute__((ext_vector_type(16))) __bf16 v16bf;
typedef __attribute__((ext_vector_type(8)))  float  v8f;

// ---------------- device helpers ----------------
__device__ __forceinline__ unsigned short f2bf(float f) {
    unsigned u = __float_as_uint(f);
    u += 0x7fffu + ((u >> 16) & 1u);          // round-to-nearest-even
    return (unsigned short)(u >> 16);
}

// monotone mapping float -> uint so unsigned max == float max
__device__ __forceinline__ unsigned fenc(float f) {
    unsigned u = __float_as_uint(f);
    return (u & 0x80000000u) ? ~u : (u | 0x80000000u);
}
__device__ __forceinline__ float fdec(unsigned u) {
    unsigned v = (u & 0x80000000u) ? (u & 0x7fffffffu) : ~u;
    return __uint_as_float(v);
}

__device__ __forceinline__ v16bf cat8(v8bf lo, v8bf hi) {
    return __builtin_shufflevector(lo, hi, 0,1,2,3,4,5,6,7,8,9,10,11,12,13,14,15);
}

// A fragment (16-bit A 16x32, ISA layout): lane holds row M=lane&15, half=lane>>4
// elem j<8  -> k = half*8 + j        (contiguous 8 -> one 16B load)
// elem j>=8 -> k = 16 + half*8 + j-8 (contiguous 8 -> one 16B load)
__device__ __forceinline__ v16bf load_fragA(const unsigned short* p, int half) {
    v8bf lo = *(const v8bf*)(p + half * 8);
    v8bf hi = *(const v8bf*)(p + 16 + half * 8);
    return cat8(lo, hi);
}
// B fragment (16-bit B 32x16, ISA layout): lane holds col N=lane&15
// elem j<8 -> k = half*8 + j, elem j>=8 -> k = 16 + half*8 + j-8 (two 16B loads)
__device__ __forceinline__ v16bf load_fragB(const unsigned short* p, int half) {
    v8bf lo = *(const v8bf*)(p + half * 8);
    v8bf hi = *(const v8bf*)(p + 16 + half * 8);
    return cat8(lo, hi);
}

// ---------------- kernels ----------------

// f32 -> bf16 elementwise
__global__ void k_cvt_bf16(const float* __restrict__ in, unsigned short* __restrict__ out, int n) {
    int i = blockIdx.x * blockDim.x + threadIdx.x;
    if (i < n) out[i] = f2bf(in[i]);
}

// W [K,C] f32 row-major -> Wt [C,K] bf16 (transposed so B fragments are contiguous in K)
__global__ void k_cvt_wt(const float* __restrict__ W, unsigned short* __restrict__ Wt, int K, int C) {
    int i = blockIdx.x * blockDim.x + threadIdx.x;
    if (i >= K * C) return;
    int k = i / C, c = i % C;
    Wt[(size_t)c * K + k] = f2bf(W[i]);
}

// C[M,Ncol] f32 = A[M,K] bf16 x B (given as Bt[Ncol,K] bf16), WMMA bf16, f32 accum.
// Each wave computes one 16-row x 32-col slab (two 16x16 N-tiles sharing A fragments):
//   per 64-K step: 2 A-loads + 4 B-loads feed 4 WMMAs (48B/lane/WMMA).
// Split-K dual accumulators per tile keep the XDL pipe free of RAW chains.
__global__ void k_gemm_wmma(const unsigned short* __restrict__ A,
                            const unsigned short* __restrict__ Bt,
                            float* __restrict__ C,
                            int M, int Ncol, int K) {
    const int wave = threadIdx.x >> 5;
    const int lane = threadIdx.x & 31;
    const int tileN0 = (blockIdx.x * 8 + wave) * 2;   // two adjacent N-tiles per wave
    const int tileM  = blockIdx.y;
    if (tileN0 * 16 >= Ncol) return;                  // wave-uniform: EXEC all-1s in WMMA

    const int half = lane >> 4;
    const int l15  = lane & 15;
    const unsigned short* arow  = A  + (size_t)(tileM * 16 + l15) * K;
    const unsigned short* brow0 = Bt + (size_t)(tileN0 * 16 + l15) * K;
    const unsigned short* brow1 = brow0 + (size_t)16 * K;

    v8f acc00 = {}, acc01 = {};   // tileN0   : split-K pair
    v8f acc10 = {}, acc11 = {};   // tileN0+1 : split-K pair
    for (int kb = 0; kb < K; kb += 64) {
        v16bf a0  = load_fragA(arow  + kb,      half);
        v16bf a1  = load_fragA(arow  + kb + 32, half);
        v16bf b00 = load_fragB(brow0 + kb,      half);
        v16bf b01 = load_fragB(brow0 + kb + 32, half);
        v16bf b10 = load_fragB(brow1 + kb,      half);
        v16bf b11 = load_fragB(brow1 + kb + 32, half);
        acc00 = __builtin_amdgcn_wmma_f32_16x16x32_bf16(false, a0, false, b00,
                                                        (short)0, acc00, false, false);
        acc10 = __builtin_amdgcn_wmma_f32_16x16x32_bf16(false, a0, false, b10,
                                                        (short)0, acc10, false, false);
        acc01 = __builtin_amdgcn_wmma_f32_16x16x32_bf16(false, a1, false, b01,
                                                        (short)0, acc01, false, false);
        acc11 = __builtin_amdgcn_wmma_f32_16x16x32_bf16(false, a1, false, b11,
                                                        (short)0, acc11, false, false);
    }

    // C/D layout: elem v -> row = tileM*16 + half*8 + v, col = tile*16 + (lane&15)
    const int ccol0 = tileN0 * 16 + l15;
    #pragma unroll
    for (int v = 0; v < 8; ++v) {
        int crow = tileM * 16 + half * 8 + v;
        float* crp = C + (size_t)crow * Ncol;
        crp[ccol0]      = acc00[v] + acc01[v];
        crp[ccol0 + 16] = acc10[v] + acc11[v];
    }
}

// per-node attention coefficients: a_s[n] = h[n,:]·asrc, a_d[n] = h[n,:]·adst
__global__ void k_attn(const float* __restrict__ h, const float* __restrict__ asrc,
                       const float* __restrict__ adst, float* __restrict__ a_s,
                       float* __restrict__ a_d, int N, int C) {
    int wave = threadIdx.x >> 5, lane = threadIdx.x & 31;
    int node = blockIdx.x * 8 + wave;
    if (node >= N) return;
    const float* hr = h + (size_t)node * C;
    float s = 0.f, d = 0.f;
    for (int c = lane; c < C; c += 32) {
        float v = hr[c];
        s += v * asrc[c];
        d += v * adst[c];
    }
    #pragma unroll
    for (int off = 16; off > 0; off >>= 1) {
        s += __shfl_down(s, off, 32);
        d += __shfl_down(d, off, 32);
    }
    if (lane == 0) { a_s[node] = s; a_d[node] = d; }
}

// zero out accumulator, init segment max/denominator
__global__ void k_layer_init(float* __restrict__ out, unsigned* __restrict__ m,
                             float* __restrict__ denom, int N, int C) {
    int i = blockIdx.x * blockDim.x + threadIdx.x;
    if (i < N * C) out[i] = 0.f;
    if (i < N) { m[i] = ENC_NEG_INF; denom[i] = 0.f; }
}

// e = leaky_relu(a_s[src]+a_d[dst]); segment max over dst (encoded u32 atomics)
__global__ void k_edge_score(const float* __restrict__ a_s, const float* __restrict__ a_d,
                             const int* __restrict__ src, const int* __restrict__ dst,
                             float* __restrict__ e, unsigned* __restrict__ m, int E) {
    int i = blockIdx.x * blockDim.x + threadIdx.x;
    if (i >= E) return;
    float v = a_s[src[i]] + a_d[dst[i]];
    v = (v > 0.f) ? v : NEGSLP * v;
    e[i] = v;
    atomicMax(&m[dst[i]], fenc(v));
}

// ex = exp(e - m[dst]); segment sum over dst
__global__ void k_edge_exp(float* __restrict__ e, const unsigned* __restrict__ m,
                           const int* __restrict__ dst, float* __restrict__ denom, int E) {
    int i = blockIdx.x * blockDim.x + threadIdx.x;
    if (i >= E) return;
    int d = dst[i];
    float ex = expf(e[i] - fdec(m[d]));
    e[i] = ex;
    atomicAdd(&denom[d], ex);
}

// out[dst,:] += alpha * h[src,:]  (thread = (edge, 4 channels))
__global__ void k_edge_scatter(const float* __restrict__ h, const float* __restrict__ ex,
                               const float* __restrict__ denom, const int* __restrict__ src,
                               const int* __restrict__ dst, float* __restrict__ out,
                               int E, int C) {
    int idx = blockIdx.x * blockDim.x + threadIdx.x;
    int c4 = C >> 2;
    int e = idx / c4;
    if (e >= E) return;
    int cq = (idx - e * c4) << 2;
    int s = src[e], d = dst[e];
    float alpha = ex[e] / (denom[d] + EPSI);
    float4 hv = *(const float4*)(h + (size_t)s * C + cq);
    float* op = out + (size_t)d * C + cq;
    atomicAdd(op + 0, alpha * hv.x);
    atomicAdd(op + 1, alpha * hv.y);
    atomicAdd(op + 2, alpha * hv.z);
    atomicAdd(op + 3, alpha * hv.w);
}

// out += bias, optional ReLU; optionally write bf16 activations for next layer's GEMM
__global__ void k_finalize(float* __restrict__ out, const float* __restrict__ bias,
                           unsigned short* __restrict__ actbf, int NC, int C, int do_relu) {
    int i = blockIdx.x * blockDim.x + threadIdx.x;
    if (i >= NC) return;
    float v = out[i] + bias[i % C];
    if (do_relu) v = fmaxf(v, 0.f);
    out[i] = v;
    if (actbf) actbf[i] = f2bf(v);
}

__global__ void k_pool_init(unsigned* __restrict__ pooled, int n) {
    int i = blockIdx.x * blockDim.x + threadIdx.x;
    if (i < n) pooled[i] = ENC_NEG_INF;
}

__global__ void k_pool_max(const float* __restrict__ out, const int* __restrict__ batch,
                           unsigned* __restrict__ pooled, int N, int C) {
    int i = blockIdx.x * blockDim.x + threadIdx.x;
    if (i >= N * C) return;
    int n = i / C, c = i - n * C;
    atomicMax(&pooled[(size_t)batch[n] * C + c], fenc(out[i]));
}

__global__ void k_final_linear(const unsigned* __restrict__ pooled,
                               const float* __restrict__ Wl, const float* __restrict__ bl,
                               float* __restrict__ y) {
    int t = threadIdx.x;
    if (t >= NG * NCLS) return;
    int g = t / NCLS, k = t - g * NCLS;
    float s = bl[k];
    for (int c = 0; c < HID; ++c)
        s += fdec(pooled[g * HID + c]) * Wl[c * NCLS + k];
    y[t] = s;
}

// ---------------- orchestration ----------------
extern "C" void kernel_launch(void* const* d_in, const int* in_sizes, int n_in,
                              void* d_out, int out_size, void* d_ws, size_t ws_size,
                              hipStream_t stream) {
    (void)in_sizes; (void)n_in; (void)out_size; (void)ws_size;

    // inputs in setup_inputs() dict order
    const float* x     = (const float*)d_in[0];
    const int*   eidx  = (const int*)d_in[1];
    const int*   src   = eidx;
    const int*   dst   = eidx + NE;
    const int*   batch = (const int*)d_in[2];
    const float* Wl[3] = {(const float*)d_in[3], (const float*)d_in[7],  (const float*)d_in[11]};
    const float* As[3] = {(const float*)d_in[4], (const float*)d_in[8],  (const float*)d_in[12]};
    const float* Ad[3] = {(const float*)d_in[5], (const float*)d_in[9],  (const float*)d_in[13]};
    const float* Bi[3] = {(const float*)d_in[6], (const float*)d_in[10], (const float*)d_in[14]};
    const float* Wlin  = (const float*)d_in[15];
    const float* blin  = (const float*)d_in[16];
    float* y = (float*)d_out;

    // workspace carve-out
    char* wp = (char*)d_ws;
    auto carve = [&](size_t bytes) -> void* {
        void* p = (void*)wp;
        wp += (bytes + 255) & ~(size_t)255;
        return p;
    };
    unsigned short* actbf  = (unsigned short*)carve((size_t)NN * FIN * 2); // layer input, bf16
    unsigned short* wtbf   = (unsigned short*)carve((size_t)FIN * HID * 2); // transposed W, bf16
    float*          hbuf   = (float*)carve((size_t)NN * HID * 4);           // h = X@W
    float*          outacc = (float*)carve((size_t)NN * HID * 4);           // aggregated output
    float*          a_s    = (float*)carve((size_t)NN * 4);
    float*          a_d    = (float*)carve((size_t)NN * 4);
    unsigned*       mmax   = (unsigned*)carve((size_t)NN * 4);
    float*          denom  = (float*)carve((size_t)NN * 4);
    float*          ebuf   = (float*)carve((size_t)NE * 4);
    unsigned*       pooled = (unsigned*)carve((size_t)NG * HID * 4);

    const int T = 256;
    const int Ks[3] = {FIN, HID, HID};

    // layer-1 input -> bf16
    {
        int n = NN * FIN;
        k_cvt_bf16<<<(n + T - 1) / T, T, 0, stream>>>(x, actbf, n);
    }

    for (int L = 0; L < 3; ++L) {
        const int K = Ks[L];

        // weight -> transposed bf16
        {
            int n = K * HID;
            k_cvt_wt<<<(n + T - 1) / T, T, 0, stream>>>(Wl[L], wtbf, K, HID);
        }
        // h = act @ W   (WMMA bf16, f32 accum; 2 N-tiles per wave)
        {
            dim3 grid(HID / (16 * 8 * 2), NN / 16);  // (2, 3125)
            k_gemm_wmma<<<grid, T, 0, stream>>>(actbf, wtbf, hbuf, NN, HID, K);
        }
        // per-node attention coefficients
        k_attn<<<(NN + 7) / 8, T, 0, stream>>>(hbuf, As[L], Ad[L], a_s, a_d, NN, HID);
        // init accumulators
        {
            int n = NN * HID;
            k_layer_init<<<(n + T - 1) / T, T, 0, stream>>>(outacc, mmax, denom, NN, HID);
        }
        // edge softmax + scatter
        k_edge_score<<<(NE + T - 1) / T, T, 0, stream>>>(a_s, a_d, src, dst, ebuf, mmax, NE);
        k_edge_exp<<<(NE + T - 1) / T, T, 0, stream>>>(ebuf, mmax, dst, denom, NE);
        {
            long long n = (long long)NE * (HID / 4);
            k_edge_scatter<<<(unsigned)((n + T - 1) / T), T, 0, stream>>>(
                hbuf, ebuf, denom, src, dst, outacc, NE, HID);
        }
        // bias (+ReLU for layers 0,1), write next-layer bf16 activations
        {
            int n = NN * HID;
            unsigned short* nxt = (L < 2) ? actbf : nullptr;
            k_finalize<<<(n + T - 1) / T, T, 0, stream>>>(outacc, Bi[L], nxt, n, HID, (L < 2) ? 1 : 0);
        }
    }

    // global max pool + classifier head
    k_pool_init<<<(NG * HID + T - 1) / T, T, 0, stream>>>(pooled, NG * HID);
    k_pool_max<<<(NN * HID + T - 1) / T, T, 0, stream>>>(outacc, batch, pooled, NN, HID);
    k_final_linear<<<1, 64, 0, stream>>>(pooled, Wlin, blin, y);
}
